// GraphConvPoolNet_8117488189556
// MI455X (gfx1250) — compile-verified, matrix operand order checked by reading, not run on hardware
//
#include <hip/hip_runtime.h>
#include <cmath>

typedef __attribute__((ext_vector_type(16))) _Float16 v16h;
typedef __attribute__((ext_vector_type(8)))  _Float16 v8h;
typedef __attribute__((ext_vector_type(4)))  _Float16 v4h;
typedef __attribute__((ext_vector_type(8)))  float    v8f;
typedef unsigned long long u64;

// counters: [0]=nNodes [1]=nEdges [2]=nClusters [3]=iterFlag [4]=always(1)

__global__ void k_init(int* cnt, int n, int e) {
    if (threadIdx.x == 0 && blockIdx.x == 0) {
        cnt[0] = n; cnt[1] = e; cnt[2] = 0; cnt[3] = 1; cnt[4] = 1;
    }
}

__global__ void k_begin_iter(int* cnt) {
    if (threadIdx.x == 0 && blockIdx.x == 0) cnt[3] = (cnt[0] > 2) ? 1 : 0;
}

__global__ void k_zero(float* __restrict__ p, long n) {
    long i = (long)blockIdx.x * blockDim.x + threadIdx.x;
    if (i < n) p[i] = 0.0f;
}

__global__ void k_fill(float* __restrict__ p, float v, long n) {
    long i = (long)blockIdx.x * blockDim.x + threadIdx.x;
    if (i < n) p[i] = v;
}

// one-time: permute f32 weights [K x N] into fragment-ordered f16:
// WT[((t*(K/32)+kb)*32 + lane)*16 + elem]  (t = n/16 tile)
__global__ void k_wconv(const float* __restrict__ W, _Float16* __restrict__ WT,
                        int K, int N) {
    long i = (long)blockIdx.x * blockDim.x + threadIdx.x;
    if (i >= (long)K * N) return;
    int k = (int)(i / N), nn = (int)(i % N);
    int t = nn >> 4, nl = nn & 15;
    int kb = k >> 5, ko = k & 31;
    int half = (ko >> 3) & 1;                 // lanes 0-15: k 0-7,16-23; 16-31: 8-15,24-31
    int lane = nl + (half << 4);
    int elem = (ko & 7) + ((ko >> 4) << 3);
    WT[(((size_t)t * (K >> 5) + kb) * 32 + lane) * 16 + elem] = (_Float16)W[i];
}

// agg[dst] += x[src]  (segment_sum of gathered rows), float4 vectorized
__global__ void k_scatter(const float* __restrict__ X, const int* __restrict__ src,
                          const int* __restrict__ dst, float* __restrict__ agg,
                          const int* __restrict__ cnt, const int* __restrict__ flag, int K) {
    if (flag[0] == 0) return;
    long tid = (long)blockIdx.x * blockDim.x + threadIdx.x;
    int K4 = K >> 2;
    int e = (int)(tid / K4), q = (int)(tid % K4);
    if (e >= cnt[1]) return;
    const float4 v = *(const float4*)(X + (size_t)src[e] * K + (size_t)q * 4);
    float* dp = agg + (size_t)dst[e] * K + (size_t)q * 4;
    atomicAdd(dp + 0, v.x); atomicAdd(dp + 1, v.y);
    atomicAdd(dp + 2, v.z); atomicAdd(dp + 3, v.w);
}

// Y = gelu(A0 @ W0 + A1 @ W1 + bias), M = cnt[0] rows, v_wmma_f32_16x16x32_f16.
// A staged in LDS (f16), fragments read as 2x ds_load_b128; B read directly from
// the fragment-ordered f16 weight tensor (2x global_load_b128 per step per lane).
template<int K, int N>
__global__ __launch_bounds__(128) void k_gemm_dual_gelu(
    const float* __restrict__ A0, const _Float16* __restrict__ W0T,
    const float* __restrict__ A1, const _Float16* __restrict__ W1T,
    const float* __restrict__ bias, float* __restrict__ Y,
    const int* __restrict__ cnt, const int* __restrict__ flag) {
    if (flag[0] == 0) return;
    const int n = cnt[0];
    const int m0 = blockIdx.y * 16;
    if (m0 >= n) return;                      // block-uniform: EXEC all-ones at WMMA
    const int n0 = blockIdx.x * 64;
    const int tid = threadIdx.x;
    const int wave = tid >> 5;
    const int lane = tid & 31;
    const int tcol = (n0 >> 4) + wave;        // this wave's 16-col tile index

    __shared__ __align__(32) _Float16 As[16][40];   // 16x32 tile, 80B row stride

    v8f acc = {};
    const int rA = tid >> 3;                  // staging: row per thread
    const int cA = (tid & 7) * 4;             // 4 consecutive k per thread
    const bool rowOk = (m0 + rA) < n;
    const int mrow = lane & 15;
    const int hi   = (lane >> 4) ? 8 : 0;

    auto pass = [&](const float* __restrict__ A, const _Float16* __restrict__ WT) {
        #pragma unroll 4
        for (int k0 = 0; k0 < K; k0 += 32) {
            float4 av = make_float4(0.0f, 0.0f, 0.0f, 0.0f);
            if (rowOk) av = *(const float4*)(A + (size_t)(m0 + rA) * K + k0 + cA);
            v16h b = *(const v16h*)(WT + (((size_t)tcol * (K >> 5) + (k0 >> 5)) * 32 + lane) * 16);
            __syncthreads();                  // prior tile fully consumed
            v4h hv;
            hv[0] = (_Float16)av.x; hv[1] = (_Float16)av.y;
            hv[2] = (_Float16)av.z; hv[3] = (_Float16)av.w;
            *(v4h*)&As[rA][cA] = hv;          // one b64 store
            __syncthreads();
            v8h alo = *(const v8h*)&As[mrow][hi];        // ds_load_b128
            v8h ahi = *(const v8h*)&As[mrow][16 + hi];   // ds_load_b128
            v16h a = __builtin_shufflevector(alo, ahi,
                     0, 1, 2, 3, 4, 5, 6, 7, 8, 9, 10, 11, 12, 13, 14, 15);
            acc = __builtin_amdgcn_wmma_f32_16x16x32_f16(false, a, false, b,
                                                         (short)0, acc, false, false);
        }
    };
    pass(A0, W0T);
    pass(A1, W1T);

    const int col   = n0 + wave * 16 + (lane & 15);
    const int rbase = m0 + ((lane >> 4) ? 8 : 0);
    #pragma unroll
    for (int r = 0; r < 8; ++r) {             // C/D layout: vgpr r -> M=r / r+8
        int row = rbase + r;
        if (row < n) {
            float v = acc[r] + bias[col];
            v = 0.5f * v * (1.0f + erff(v * 0.70710678118654752f));
            Y[(size_t)row * N + col] = v;
        }
    }
}

// raw[e] = dot(x[src], w[:d]) + dot(x[dst], w[d:]) + b   (one wave32 per edge)
__global__ void k_edge_score(const float* __restrict__ X, const int* __restrict__ src,
                             const int* __restrict__ dst, const float* __restrict__ w,
                             const float* __restrict__ bsc, float* __restrict__ raw,
                             const int* __restrict__ cnt, const int* __restrict__ flag, int d) {
    if (flag[0] == 0) return;
    int gid = blockIdx.x * blockDim.x + threadIdx.x;
    int e = gid >> 5, lane = gid & 31;
    if (e >= cnt[1]) return;
    int s = src[e], t = dst[e];
    float acc = 0.0f;
    for (int f = lane; f < d; f += 32)
        acc += X[(size_t)s * d + f] * w[f] + X[(size_t)t * d + f] * w[d + f];
    for (int off = 16; off > 0; off >>= 1) acc += __shfl_xor(acc, off, 32);
    if (lane == 0) raw[e] = acc + bsc[0];
}

__device__ inline void atomicMaxFloat(float* addr, float val) {
    unsigned int* ua = (unsigned int*)addr;
    unsigned int old = *ua;
    while (__uint_as_float(old) < val) {
        unsigned int assumed = old;
        old = atomicCAS(ua, assumed, __float_as_uint(val));
        if (old == assumed) break;
    }
}

__global__ void k_segmax(const float* __restrict__ raw, const int* __restrict__ dst,
                         float* __restrict__ m, const int* __restrict__ cnt,
                         const int* __restrict__ flag) {
    if (flag[0] == 0) return;
    int e = blockIdx.x * blockDim.x + threadIdx.x;
    if (e >= cnt[1]) return;
    atomicMaxFloat(&m[dst[e]], raw[e]);
}

__global__ void k_fixm(float* __restrict__ m, const int* __restrict__ cnt,
                       const int* __restrict__ flag) {
    if (flag[0] == 0) return;
    int i = blockIdx.x * blockDim.x + threadIdx.x;
    if (i >= cnt[0]) return;
    if (!__builtin_isfinite(m[i])) m[i] = 0.0f;
}

__global__ void k_exp(const float* __restrict__ raw, const int* __restrict__ dst,
                      const float* __restrict__ m, float* __restrict__ ex,
                      float* __restrict__ den, const int* __restrict__ cnt,
                      const int* __restrict__ flag) {
    if (flag[0] == 0) return;
    int e = blockIdx.x * blockDim.x + threadIdx.x;
    if (e >= cnt[1]) return;
    float v = expf(raw[e] - m[dst[e]]);
    ex[e] = v;
    atomicAdd(&den[dst[e]], v);
}

__global__ void k_score_final(float* __restrict__ ex, const int* __restrict__ dst,
                              const float* __restrict__ den, const int* __restrict__ cnt,
                              const int* __restrict__ flag) {
    if (flag[0] == 0) return;
    int e = blockIdx.x * blockDim.x + threadIdx.x;
    if (e >= cnt[1]) return;
    ex[e] = ex[e] / (den[dst[e]] + 1e-16f) + 0.5f;   // add_to_edge_score
}

// sort key: descending score, ascending index (stable argsort(-score))
__global__ void k_keys_score(const float* __restrict__ sc, u64* __restrict__ keys,
                             const int* __restrict__ cnt, const int* __restrict__ flag,
                             int npad) {
    if (flag[0] == 0) return;
    int i = blockIdx.x * blockDim.x + threadIdx.x;
    if (i >= npad) return;
    if (i < cnt[1]) {
        unsigned int b = __float_as_uint(sc[i]);
        b = (b & 0x80000000u) ? ~b : (b | 0x80000000u);   // orderable
        keys[i] = ((u64)(~b) << 32) | (unsigned int)i;
    } else {
        keys[i] = ~0ull;
    }
}

__global__ __launch_bounds__(1024) void k_bitonic(u64* __restrict__ keys, int npad,
                                                  const int* __restrict__ flag) {
    if (flag[0] == 0) return;
    const int tid = threadIdx.x;
    for (int k = 2; k <= npad; k <<= 1)
        for (int j = k >> 1; j > 0; j >>= 1) {
            for (int i = tid; i < npad; i += 1024) {
                int l = i ^ j;
                if (l > i) {
                    u64 a = keys[i], b = keys[l];
                    bool asc = ((i & k) == 0);
                    if ((a > b) == asc) { keys[i] = b; keys[l] = a; }
                }
            }
            __syncthreads();
        }
}

__global__ void k_greedy(const u64* __restrict__ keys, const int* __restrict__ src,
                         const int* __restrict__ dst, int* __restrict__ rem,
                         int* __restrict__ cl, int* __restrict__ eo,
                         int* __restrict__ cnt, const int* __restrict__ flag) {
    if (flag[0] == 0) return;
    if (threadIdx.x != 0 || blockIdx.x != 0) return;
    int n = cnt[0], E = cnt[1];
    for (int i = 0; i < n; ++i) rem[i] = 1;
    int c = 0;
    for (int kk = 0; kk < E; ++kk) {
        int e = (int)(keys[kk] & 0xffffffffu);
        int s = src[e]; if (!rem[s]) continue;
        int t = dst[e]; if (!rem[t]) continue;
        rem[s] = 0; rem[t] = 0;
        cl[s] = c; cl[t] = c; eo[c] = e; ++c;
    }
    for (int i = 0; i < n; ++i) if (rem[i]) { cl[i] = c; eo[c] = -1; ++c; }
    cnt[2] = c;
}

__global__ void k_pool_sum(const float* __restrict__ X, const int* __restrict__ cl,
                           float* __restrict__ xn, const int* __restrict__ cnt,
                           const int* __restrict__ flag, int d) {
    if (flag[0] == 0) return;
    long tid = (long)blockIdx.x * blockDim.x + threadIdx.x;
    int i = (int)(tid / d), f = (int)(tid % d);
    if (i >= cnt[0]) return;
    atomicAdd(&xn[(size_t)cl[i] * d + f], X[(size_t)i * d + f]);
}

__global__ void k_pool_mult(float* __restrict__ xn, const int* __restrict__ eo,
                            const float* __restrict__ sc, const int* __restrict__ cnt,
                            const int* __restrict__ flag, int d) {
    if (flag[0] == 0) return;
    long tid = (long)blockIdx.x * blockDim.x + threadIdx.x;
    int ci = (int)(tid / d), f = (int)(tid % d);
    if (ci >= cnt[2]) return;
    int e = eo[ci];
    float m = (e >= 0) ? sc[e] : 1.0f;
    xn[(size_t)ci * d + f] *= m;
}

__global__ void k_remap(const int* __restrict__ src, const int* __restrict__ dst,
                        const int* __restrict__ cl, u64* __restrict__ keys,
                        const int* __restrict__ cnt, const int* __restrict__ flag,
                        int npad) {
    if (flag[0] == 0) return;
    int i = blockIdx.x * blockDim.x + threadIdx.x;
    if (i >= npad) return;
    if (i < cnt[1]) {
        u64 c = (u64)cnt[2];
        keys[i] = (u64)cl[src[i]] * c + (u64)cl[dst[i]];
    } else {
        keys[i] = ~0ull;
    }
}

__global__ void k_dedupe(const u64* __restrict__ keys, int* __restrict__ sB,
                         int* __restrict__ dB, int* __restrict__ cnt,
                         const int* __restrict__ flag) {
    if (flag[0] == 0) return;
    if (threadIdx.x != 0 || blockIdx.x != 0) return;
    u64 c = (u64)cnt[2];
    int E = cnt[1];
    int m = 0; u64 prev = ~0ull;
    for (int kk = 0; kk < E; ++kk) {
        u64 k = keys[kk];
        if (k == ~0ull) break;
        if (kk == 0 || k != prev) { sB[m] = (int)(k / c); dB[m] = (int)(k % c); ++m; prev = k; }
    }
    cnt[1] = m; cnt[0] = (int)c;
}

__global__ void k_commit(const int* __restrict__ sB, const int* __restrict__ dB,
                         int* __restrict__ sA, int* __restrict__ dA,
                         const int* __restrict__ cnt, const int* __restrict__ flag) {
    if (flag[0] == 0) return;
    int e = blockIdx.x * blockDim.x + threadIdx.x;
    if (e >= cnt[1]) return;
    sA[e] = sB[e]; dA[e] = dB[e];
}

// readout: segment_max over i%2 then dot with w_out, + b_out
__global__ void k_readout(const float* __restrict__ X, const float* __restrict__ wout,
                          const float* __restrict__ bout, float* __restrict__ out,
                          const int* __restrict__ cnt, int d) {
    __shared__ float red[256];
    int tid = threadIdx.x;
    int n = cnt[0];
    float part = 0.0f;
    for (int f = tid; f < d; f += 256) {
        float m0 = -3.402823466e38f, m1 = -3.402823466e38f;
        for (int i = 0; i < n; ++i) {
            float v = X[(size_t)i * d + f];
            if ((i & 1) == 0) m0 = fmaxf(m0, v); else m1 = fmaxf(m1, v);
        }
        part += m0 * wout[f] + m1 * wout[d + f];
    }
    red[tid] = part;
    __syncthreads();
    for (int s = 128; s > 0; s >>= 1) { if (tid < s) red[tid] += red[tid + s]; __syncthreads(); }
    if (tid == 0) out[0] = red[0] + bout[0];
}

extern "C" void kernel_launch(void* const* d_in, const int* in_sizes, int n_in,
                              void* d_out, int out_size, void* d_ws, size_t ws_size,
                              hipStream_t stream) {
    const float* x_in  = (const float*)d_in[0];
    const int*   src_i = (const int*)d_in[1];
    const int*   dst_i = (const int*)d_in[2];
    const float *w_in  = (const float*)d_in[3],  *b_in = (const float*)d_in[4],
                *r_in  = (const float*)d_in[5];
    const float *w_c1  = (const float*)d_in[6],  *b_c1 = (const float*)d_in[7],
                *r_c1  = (const float*)d_in[8];
    const float *w_p1  = (const float*)d_in[9],  *b_p1 = (const float*)d_in[10];
    const float *w_c2  = (const float*)d_in[11], *b_c2 = (const float*)d_in[12],
                *r_c2  = (const float*)d_in[13];
    const float *w_p2  = (const float*)d_in[14], *b_p2 = (const float*)d_in[15];
    const float *w_c3  = (const float*)d_in[16], *b_c3 = (const float*)d_in[17],
                *r_c3  = (const float*)d_in[18];
    const float *w_pl  = (const float*)d_in[19], *b_pl = (const float*)d_in[20];
    const float *w_cl  = (const float*)d_in[21], *b_cl = (const float*)d_in[22],
                *r_cl  = (const float*)d_in[23];
    const float *w_out = (const float*)d_in[24], *b_out = (const float*)d_in[25];

    const int FIN = 32, D3 = 768;
    const int NMAX = in_sizes[0] / FIN;       // 8192
    const int E    = in_sizes[1];             // 98304
    int Epad = 1; while (Epad < E) Epad <<= 1;

    unsigned char* base = (unsigned char*)d_ws;
    size_t off = 0;
    auto alloc = [&](size_t bytes) -> void* {
        void* p = base + off;
        off = (off + bytes + 255) & ~(size_t)255;
        return p;
    };
    float* X0   = (float*)alloc((size_t)NMAX * D3 * 4);
    float* X1   = (float*)alloc((size_t)NMAX * D3 * 4);
    float* AGG  = (float*)alloc((size_t)NMAX * D3 * 4);
    float* RAW  = (float*)alloc((size_t)E * 4);
    float* SC   = (float*)alloc((size_t)E * 4);
    float* MN   = (float*)alloc((size_t)NMAX * 4);
    float* DEN  = (float*)alloc((size_t)NMAX * 4);
    u64*   KEYS = (u64*)alloc((size_t)Epad * 8);
    int*   CL   = (int*)alloc((size_t)NMAX * 4);
    int*   EO   = (int*)alloc((size_t)NMAX * 4);
    int*   REM  = (int*)alloc((size_t)NMAX * 4);
    int*   SA   = (int*)alloc((size_t)E * 4);
    int*   DA   = (int*)alloc((size_t)E * 4);
    int*   SB   = (int*)alloc((size_t)E * 4);
    int*   DB   = (int*)alloc((size_t)E * 4);
    _Float16* WTin = (_Float16*)alloc((size_t)32  * 64  * 2);
    _Float16* RTin = (_Float16*)alloc((size_t)32  * 64  * 2);
    _Float16* WTc1 = (_Float16*)alloc((size_t)64  * 256 * 2);
    _Float16* RTc1 = (_Float16*)alloc((size_t)64  * 256 * 2);
    _Float16* WTc2 = (_Float16*)alloc((size_t)256 * 512 * 2);
    _Float16* RTc2 = (_Float16*)alloc((size_t)256 * 512 * 2);
    _Float16* WTc3 = (_Float16*)alloc((size_t)512 * 768 * 2);
    _Float16* RTc3 = (_Float16*)alloc((size_t)512 * 768 * 2);
    _Float16* WTcl = (_Float16*)alloc((size_t)768 * 768 * 2);
    _Float16* RTcl = (_Float16*)alloc((size_t)768 * 768 * 2);
    int*   CNT  = (int*)alloc(64);

    hipMemcpyAsync(SA, src_i, (size_t)E * 4, hipMemcpyDeviceToDevice, stream);
    hipMemcpyAsync(DA, dst_i, (size_t)E * 4, hipMemcpyDeviceToDevice, stream);
    k_init<<<1, 1, 0, stream>>>(CNT, NMAX, E);
    const int* ALWAYS = CNT + 4;
    const int* ITER   = CNT + 3;

    auto wconv = [&](const float* W, _Float16* WT, int K, int N) {
        long t = (long)K * N;
        k_wconv<<<(unsigned)((t + 255) / 256), 256, 0, stream>>>(W, WT, K, N);
    };
    wconv(w_in, WTin, 32, 64);   wconv(r_in, RTin, 32, 64);
    wconv(w_c1, WTc1, 64, 256);  wconv(r_c1, RTc1, 64, 256);
    wconv(w_c2, WTc2, 256, 512); wconv(r_c2, RTc2, 256, 512);
    wconv(w_c3, WTc3, 512, 768); wconv(r_c3, RTc3, 512, 768);
    wconv(w_cl, WTcl, 768, 768); wconv(r_cl, RTcl, 768, 768);

    auto scatter = [&](const float* Xin, int K, const int* flag) {
        long zn = (long)NMAX * K;
        k_zero<<<(unsigned)((zn + 255) / 256), 256, 0, stream>>>(AGG, zn);
        long sn = (long)E * (K >> 2);
        k_scatter<<<(unsigned)((sn + 255) / 256), 256, 0, stream>>>(Xin, SA, DA, AGG, CNT, flag, K);
    };

    auto pool = [&](const float* Xin, float* Xout, const float* wp, const float* bp,
                    int d, const int* flag) {
        k_edge_score<<<(E + 7) / 8, 256, 0, stream>>>(Xin, SA, DA, wp, bp, RAW, CNT, flag, d);
        k_fill<<<(NMAX + 255) / 256, 256, 0, stream>>>(MN, -INFINITY, NMAX);
        k_segmax<<<(E + 255) / 256, 256, 0, stream>>>(RAW, DA, MN, CNT, flag);
        k_fixm<<<(NMAX + 255) / 256, 256, 0, stream>>>(MN, CNT, flag);
        k_fill<<<(NMAX + 255) / 256, 256, 0, stream>>>(DEN, 0.0f, NMAX);
        k_exp<<<(E + 255) / 256, 256, 0, stream>>>(RAW, DA, MN, SC, DEN, CNT, flag);
        k_score_final<<<(E + 255) / 256, 256, 0, stream>>>(SC, DA, DEN, CNT, flag);
        k_keys_score<<<(Epad + 255) / 256, 256, 0, stream>>>(SC, KEYS, CNT, flag, Epad);
        k_bitonic<<<1, 1024, 0, stream>>>(KEYS, Epad, flag);
        k_greedy<<<1, 1, 0, stream>>>(KEYS, SA, DA, REM, CL, EO, CNT, flag);
        long zn = (long)NMAX * d;
        k_zero<<<(unsigned)((zn + 255) / 256), 256, 0, stream>>>(Xout, zn);
        k_pool_sum<<<(unsigned)((zn + 255) / 256), 256, 0, stream>>>(Xin, CL, Xout, CNT, flag, d);
        k_pool_mult<<<(unsigned)((zn + 255) / 256), 256, 0, stream>>>(Xout, EO, SC, CNT, flag, d);
        k_remap<<<(Epad + 255) / 256, 256, 0, stream>>>(SA, DA, CL, KEYS, CNT, flag, Epad);
        k_bitonic<<<1, 1024, 0, stream>>>(KEYS, Epad, flag);
        k_dedupe<<<1, 1, 0, stream>>>(KEYS, SB, DB, CNT, flag);
        k_commit<<<(E + 255) / 256, 256, 0, stream>>>(SB, DB, SA, DA, CNT, flag);
    };

    const unsigned MT = (NMAX + 15) / 16;

    // fixed, deterministic launch sequence (graph-capturable)
    scatter(x_in, 32, ALWAYS);
    k_gemm_dual_gelu<32, 64><<<dim3(1, MT), 128, 0, stream>>>(AGG, WTin, x_in, RTin, b_in, X1, CNT, ALWAYS);
    scatter(X1, 64, ALWAYS);
    k_gemm_dual_gelu<64, 256><<<dim3(4, MT), 128, 0, stream>>>(AGG, WTc1, X1, RTc1, b_c1, X0, CNT, ALWAYS);
    pool(X0, X1, w_p1, b_p1, 256, ALWAYS);
    scatter(X1, 256, ALWAYS);
    k_gemm_dual_gelu<256, 512><<<dim3(8, MT), 128, 0, stream>>>(AGG, WTc2, X1, RTc2, b_c2, X0, CNT, ALWAYS);
    pool(X0, X1, w_p2, b_p2, 512, ALWAYS);
    scatter(X1, 512, ALWAYS);
    k_gemm_dual_gelu<512, 768><<<dim3(12, MT), 128, 0, stream>>>(AGG, WTc3, X1, RTc3, b_c3, X0, CNT, ALWAYS);
    for (int it = 0; it < 16; ++it) {
        k_begin_iter<<<1, 1, 0, stream>>>(CNT);
        pool(X0, X1, w_pl, b_pl, 768, ITER);     // skipped (flag=0) once n<=2; data stays in X0
        scatter(X1, 768, ITER);
        k_gemm_dual_gelu<768, 768><<<dim3(12, MT), 128, 0, stream>>>(AGG, WTcl, X1, RTcl, b_cl, X0, CNT, ITER);
    }
    k_readout<<<1, 256, 0, stream>>>(X0, w_out, b_out, (float*)d_out, CNT, 768);

    (void)n_in; (void)out_size; (void)ws_size;
}
